// E3Decoder_21792664060155
// MI455X (gfx1250) — matrix operand order repaired and checked by reading, not compiled, not run on hardware
//
#include <hip/hip_runtime.h>

// ---------------- types ----------------
typedef __attribute__((ext_vector_type(16))) _Float16 v16h;
typedef __attribute__((ext_vector_type(8)))  _Float16 v8h;
typedef __attribute__((ext_vector_type(8)))  float    v8f;
typedef __attribute__((ext_vector_type(4)))  float    v4f;
typedef __attribute__((ext_vector_type(4)))  int      v4i;

typedef __attribute__((address_space(1))) v4i* gbl_v4i_ptr;  // global
typedef __attribute__((address_space(3))) v4i* lds_v4i_ptr;  // LDS

#define NN    384          // nodes
#define HH    128          // hidden
#define RR    16           // rbf
#define LL    3            // layers
#define NTJ   (NN/16)      // 24 j-tiles
#define WPB   4            // waves per edge-block
#define CUT2  144.0f
#define GAMMA 1.77777778f  // (R/CUTOFF)^2
#define MUSTEP 0.8f        // CUTOFF/(R-1)

// -------- CDNA5 async global->LDS staging (guarded; sync fallback) --------
#if defined(__has_builtin)
#if __has_builtin(__builtin_amdgcn_global_load_async_to_lds_b128)
#define USE_ASYNC_LDS 1
#endif
#endif
#ifndef USE_ASYNC_LDS
#define USE_ASYNC_LDS 0
#endif

__device__ __forceinline__ void copy16_lds(void* lds, const void* g) {
#if USE_ASYNC_LDS
  __builtin_amdgcn_global_load_async_to_lds_b128(
      (gbl_v4i_ptr)(__UINTPTR_TYPE__)g, (lds_v4i_ptr)(__UINTPTR_TYPE__)lds, 0, 0);
#else
  *(v8h*)lds = *(const v8h*)g;
#endif
}

__device__ __forceinline__ void wait_async_lds() {
#if USE_ASYNC_LDS
#if defined(__has_builtin) && __has_builtin(__builtin_amdgcn_s_wait_asynccnt)
  __builtin_amdgcn_s_wait_asynccnt(0);
#else
  asm volatile("s_wait_asynccnt 0" ::: "memory");
#endif
#endif
}

__device__ __forceinline__ void lds_fence() {
#if defined(__has_builtin) && __has_builtin(__builtin_amdgcn_s_wait_dscnt)
  __builtin_amdgcn_s_wait_dscnt(0);
#else
  asm volatile("s_wait_dscnt 0" ::: "memory");
#endif
}

__device__ __forceinline__ float silu_f(float v) {
  return v / (1.0f + __expf(-v));
}

__device__ __forceinline__ v8f wmma_f16(v16h a, v16h b, v8f c) {
  return __builtin_amdgcn_wmma_f32_16x16x32_f16(false, a, false, b, (short)0, c, false, false);
}

// ---------------- weight prep: f32 row-major (K x N) -> f16 B-fragment tiles ----
// Block (kt,nt) of 32x16 stored as 512 f16: offset = ((((k>>3)&1) ? 16:0)+(n&15))*16 + ((k&16)>>1) + (k&7)
__global__ void prep_b_kernel(const float* __restrict__ src, int srcK, int ld,
                              _Float16* __restrict__ dst, int Kpad, int Nw) {
  int total = Kpad * Nw;
  int ntiles = Nw >> 4;
  for (int idx = blockIdx.x * blockDim.x + threadIdx.x; idx < total;
       idx += gridDim.x * blockDim.x) {
    int k = idx / Nw, n = idx - k * Nw;
    float v = (k < srcK) ? src[k * ld + n] : 0.0f;
    int kt = k >> 5, nt = n >> 4;
    int kin = k & 31, hi = (kin >> 3) & 1;
    int off = (kt * ntiles + nt) * 512 + ((hi << 4) + (n & 15)) * 16 + ((kin >> 4) << 3) + (kin & 7);
    dst[off] = (_Float16)v;
  }
}

// ---------------- center coords ----------------
__global__ void center_kernel(const float* __restrict__ anchor, float* __restrict__ x) {
  __shared__ float mean[3];
  int tid = threadIdx.x;
  if (tid < 3) {
    float s = 0.0f;
    for (int r = 0; r < NN; ++r) s += anchor[r * 3 + tid];
    mean[tid] = s * (1.0f / (float)NN);
  }
  __syncthreads();
  if (tid < NN) {
    x[tid * 3 + 0] = anchor[tid * 3 + 0] - mean[0];
    x[tid * 3 + 1] = anchor[tid * 3 + 1] - mean[1];
    x[tid * 3 + 2] = anchor[tid * 3 + 2] - mean[2];
  }
}

// ---------------- generic wave-per-tile WMMA GEMM ----------------
__global__ void __launch_bounds__(32)
gemm16_kernel(const float* __restrict__ A0, const float* __restrict__ A1,
              int K0, int Ktot, int ldA,
              const _Float16* __restrict__ Bp, const float* __restrict__ bias,
              const float* __restrict__ resid, float* __restrict__ C,
              int Nw, int act) {
  int ntiles = Nw >> 4;
  int mt = blockIdx.x / ntiles;
  int nt = blockIdx.x - mt * ntiles;
  int lane = threadIdx.x & 31;
  int rlo = lane & 15, hi = lane >> 4;
  int row = mt * 16 + rlo;

  v8f acc = {};
  int ktn = Ktot >> 5;
  for (int kt = 0; kt < ktn; ++kt) {
    int kb = kt << 5;
    const float* ap = (kb < K0) ? (A0 + row * ldA + kb)
                                : (A1 + row * ldA + (kb - K0));
    v16h a;
#pragma unroll
    for (int g = 0; g < 2; ++g) {
      const v4f* p = (const v4f*)(ap + g * 16 + hi * 8);
      v4f f0 = p[0];
      v4f f1 = p[1];
#pragma unroll
      for (int t = 0; t < 4; ++t) {
        a[g * 8 + t]     = (_Float16)f0[t];
        a[g * 8 + 4 + t] = (_Float16)f1[t];
      }
    }
    v16h b = *(const v16h*)(Bp + (kt * ntiles + nt) * 512 + lane * 16);
    acc = wmma_f16(a, b, acc);
  }

  int col = nt * 16 + rlo;
  float bv = bias ? bias[col] : 0.0f;
#pragma unroll
  for (int v = 0; v < 8; ++v) {
    int r = mt * 16 + hi * 8 + v;
    float val = acc[v] + bv;
    if (act) val = silu_f(val);
    if (resid) val += resid[r * Nw + col];
    C[r * Nw + col] = val;
  }
}

// ---------------- edge pipeline: one block per receiver i, 4 waves sweep j-tiles ----
__global__ void __launch_bounds__(128)
edge_kernel(const float* __restrict__ x, const float* __restrict__ P,
            const float* __restrict__ Q,
            const _Float16* __restrict__ Wc, const _Float16* __restrict__ We2,
            const _Float16* __restrict__ Wc1,
            const float* __restrict__ eb2, const float* __restrict__ cb1,
            const float* __restrict__ cw2,
            float* __restrict__ pagg, float* __restrict__ pcoord) {
  // weights staged once per block, reused by all 24 j-tiles
  __shared__ _Float16 sWc[32 * 128];     //  8 KB
  __shared__ _Float16 sWe2[128 * 128];   // 32 KB
  __shared__ _Float16 sWc1[128 * 128];   // 32 KB
  __shared__ float    sP[128];
  // per-wave pipeline buffers
  __shared__ float    sQ[WPB][16 * 128];   // 32 KB
  __shared__ _Float16 sE[WPB][16 * 32];
  __shared__ _Float16 sM1[WPB][16 * 128];  // also reused as sT in stage 3/4
  __shared__ _Float16 sM[WPB][16 * 128];
  __shared__ float    sMask[WPB][16];
  __shared__ float    sDiff[WPB][16 * 3];

  int i = blockIdx.x;
  int tid = threadIdx.x;
  int wv = tid >> 5;
  int lane = tid & 31;
  int rlo = lane & 15, hi = lane >> 4;

  // ---- cooperative weight + P staging (async global->LDS) ----
  for (int c = tid; c < 512; c += 128)
    copy16_lds((char*)sWc + c * 16, (const char*)Wc + c * 16);
  for (int c = tid; c < 2048; c += 128)
    copy16_lds((char*)sWe2 + c * 16, (const char*)We2 + c * 16);
  for (int c = tid; c < 2048; c += 128)
    copy16_lds((char*)sWc1 + c * 16, (const char*)Wc1 + c * 16);
  if (tid < 32)
    copy16_lds((char*)sP + tid * 16, (const char*)(P + i * 128) + tid * 16);
  wait_async_lds();     // each wave drains its own async ops
  __syncthreads();      // then weights visible to all waves

  float xi0 = x[i * 3 + 0], xi1 = x[i * 3 + 1], xi2 = x[i * 3 + 2];

  _Float16* sTp = sM1[wv];  // stage-3 output aliases the dead m1 buffer

  for (int jt = wv; jt < NTJ; jt += WPB) {
    int j0 = jt * 16;
    lds_fence();  // prior iteration's LDS reads done before re-staging sQ

    // ---- stage 0: Q tile (async), geometry, mask, rbf ----
    {
      const char* qsrc = (const char*)(Q + j0 * 128);
      char* qdst = (char*)sQ[wv];
#pragma unroll
      for (int c = 0; c < 16; ++c)
        copy16_lds(qdst + (lane + c * 32) * 16, qsrc + (lane + c * 32) * 16);
    }
    if (lane < 16) {
      int j = j0 + lane;
      float dx = xi0 - x[j * 3 + 0];
      float dy = xi1 - x[j * 3 + 1];
      float dz = xi2 - x[j * 3 + 2];
      float d2 = dx * dx + dy * dy + dz * dz;
      int dij = i - j; if (dij < 0) dij = -dij;
      bool mb = ((d2 < CUT2) && (j != i)) || (dij == 1);
      float mf = mb ? 1.0f : 0.0f;
      sMask[wv][lane] = mf;
      sDiff[wv][lane * 3 + 0] = dx;
      sDiff[wv][lane * 3 + 1] = dy;
      sDiff[wv][lane * 3 + 2] = dz;
      float d = sqrtf(mb ? d2 : 1.0f) * mf;
#pragma unroll
      for (int r = 0; r < RR; ++r) {
        float dm = d - MUSTEP * (float)r;
        sE[wv][lane * 32 + r] = (_Float16)__expf(-GAMMA * dm * dm);
      }
#pragma unroll
      for (int r = RR; r < 32; ++r) sE[wv][lane * 32 + r] = (_Float16)0.0f;
    }
    wait_async_lds();
    lds_fence();

    // ---- stage 1: m1 = silu(e @ Wc + P_i + Q_j) ----
    v16h ae;
    {
      const _Float16* eb = sE[wv] + rlo * 32 + hi * 8;
      v8h e0 = *(const v8h*)(eb);
      v8h e1 = *(const v8h*)(eb + 16);
#pragma unroll
      for (int t = 0; t < 8; ++t) { ae[t] = e0[t]; ae[8 + t] = e1[t]; }
    }
#pragma unroll
    for (int nt = 0; nt < 8; ++nt) {
      v16h b = *(const v16h*)(sWc + nt * 512 + lane * 16);
      v8f c = {};
      c = wmma_f16(ae, b, c);
      int ch = nt * 16 + rlo;
      float pv = sP[ch];
#pragma unroll
      for (int v = 0; v < 8; ++v) {
        int er = hi * 8 + v;
        float val = silu_f(c[v] + pv + sQ[wv][er * 128 + ch]);
        sM1[wv][er * 128 + ch] = (_Float16)val;
      }
    }
    lds_fence();

    // ---- stage 2: m = silu(m1 @ eW2 + eb2) * mask ; partial agg ----
    v16h am[4];
#pragma unroll
    for (int kt = 0; kt < 4; ++kt) {
      const _Float16* p = sM1[wv] + rlo * 128 + kt * 32 + hi * 8;
      v8h a0 = *(const v8h*)(p);
      v8h a1 = *(const v8h*)(p + 16);
#pragma unroll
      for (int t = 0; t < 8; ++t) { am[kt][t] = a0[t]; am[kt][8 + t] = a1[t]; }
    }
#pragma unroll
    for (int nt = 0; nt < 8; ++nt) {
      v8f c = {};
#pragma unroll
      for (int kt = 0; kt < 4; ++kt) {
        v16h b = *(const v16h*)(sWe2 + (kt * 8 + nt) * 512 + lane * 16);
        c = wmma_f16(am[kt], b, c);
      }
      int ch = nt * 16 + rlo;
      float bb = eb2[ch];
      float s = 0.0f;
#pragma unroll
      for (int v = 0; v < 8; ++v) {
        int er = hi * 8 + v;
        float val = silu_f(c[v] + bb) * sMask[wv][er];
        sM[wv][er * 128 + ch] = (_Float16)val;
        s += val;
      }
      s += __shfl_xor(s, 16);
      if (hi == 0) pagg[(i * NTJ + jt) * 128 + ch] = s;
    }
    lds_fence();

    // ---- stage 3: t = silu(m @ cW1 + cb1)  (written over dead m1 buffer) ----
#pragma unroll
    for (int kt = 0; kt < 4; ++kt) {
      const _Float16* p = sM[wv] + rlo * 128 + kt * 32 + hi * 8;
      v8h a0 = *(const v8h*)(p);
      v8h a1 = *(const v8h*)(p + 16);
#pragma unroll
      for (int t = 0; t < 8; ++t) { am[kt][t] = a0[t]; am[kt][8 + t] = a1[t]; }
    }
#pragma unroll
    for (int nt = 0; nt < 8; ++nt) {
      v8f c = {};
#pragma unroll
      for (int kt = 0; kt < 4; ++kt) {
        v16h b = *(const v16h*)(sWc1 + (kt * 8 + nt) * 512 + lane * 16);
        c = wmma_f16(am[kt], b, c);
      }
      int ch = nt * 16 + rlo;
      float bb = cb1[ch];
#pragma unroll
      for (int v = 0; v < 8; ++v) {
        int er = hi * 8 + v;
        sTp[er * 128 + ch] = (_Float16)silu_f(c[v] + bb);
      }
    }
    lds_fence();

    // ---- stage 4: w = (t @ cW2) * mask ; partial coord update ----
    float wsum = 0.0f;
    const _Float16* trow = sTp + rlo * 128 + hi * 64;
#pragma unroll
    for (int b8 = 0; b8 < 8; ++b8) {
      v8h tv = *(const v8h*)(trow + b8 * 8);
#pragma unroll
      for (int t = 0; t < 8; ++t) wsum += (float)tv[t] * cw2[hi * 64 + b8 * 8 + t];
    }
    wsum += __shfl_xor(wsum, 16);
    float w = wsum * sMask[wv][rlo];
    float cx = 0.0f, cy = 0.0f, cz = 0.0f;
    if (hi == 0) {
      cx = sDiff[wv][rlo * 3 + 0] * w;
      cy = sDiff[wv][rlo * 3 + 1] * w;
      cz = sDiff[wv][rlo * 3 + 2] * w;
    }
#pragma unroll
    for (int off = 1; off < 32; off <<= 1) {
      cx += __shfl_xor(cx, off);
      cy += __shfl_xor(cy, off);
      cz += __shfl_xor(cz, off);
    }
    if (lane == 0) {
      pcoord[(i * NTJ + jt) * 3 + 0] = cx;
      pcoord[(i * NTJ + jt) * 3 + 1] = cy;
      pcoord[(i * NTJ + jt) * 3 + 2] = cz;
    }
  }
}

// ---------------- deterministic tile reduction: agg + coord/x update ----------
__global__ void reduce_kernel(const float* __restrict__ pagg,
                              const float* __restrict__ pcoord,
                              float* __restrict__ agg, float* __restrict__ x) {
  int idx = blockIdx.x * blockDim.x + threadIdx.x;
  if (idx < NN * HH) {
    int i = idx >> 7, ch = idx & 127;
    float s = 0.0f;
    for (int jt = 0; jt < NTJ; ++jt) s += pagg[(i * NTJ + jt) * 128 + ch];
    agg[idx] = s;
  } else if (idx < NN * HH + NN * 3) {
    int t = idx - NN * HH;
    int i = t / 3, c = t - i * 3;
    float s = 0.0f;
    for (int jt = 0; jt < NTJ; ++jt) s += pcoord[(i * NTJ + jt) * 3 + c];
    x[t] += s;
  }
}

// ---------------- host orchestration ----------------
extern "C" void kernel_launch(void* const* d_in, const int* in_sizes, int n_in,
                              void* d_out, int out_size, void* d_ws, size_t ws_size,
                              hipStream_t stream) {
  (void)in_sizes; (void)n_in; (void)out_size; (void)ws_size;
  const float* z       = (const float*)d_in[0];
  const float* anchor  = (const float*)d_in[1];
  const float* proj_W  = (const float*)d_in[2];
  const float* proj_b  = (const float*)d_in[3];
  const float* eW1     = (const float*)d_in[4];
  const float* eb1     = (const float*)d_in[5];
  const float* eW2     = (const float*)d_in[6];
  const float* eb2     = (const float*)d_in[7];
  const float* nW1     = (const float*)d_in[8];
  const float* nb1     = (const float*)d_in[9];
  const float* nW2     = (const float*)d_in[10];
  const float* nb2     = (const float*)d_in[11];
  const float* cW1     = (const float*)d_in[12];
  const float* cb1     = (const float*)d_in[13];
  const float* cW2     = (const float*)d_in[14];

  char* base = (char*)d_ws;
  size_t off = 0;
  auto carve = [&](size_t bytes) -> void* {
    off = (off + 255) & ~(size_t)255;
    void* p = base + off;
    off += bytes;
    return p;
  };
  float* xw   = (float*)carve(NN * 3 * sizeof(float));
  float* h    = (float*)carve(NN * HH * sizeof(float));
  float* Pm   = (float*)carve(NN * HH * sizeof(float));
  float* Qm   = (float*)carve(NN * HH * sizeof(float));
  float* agg  = (float*)carve(NN * HH * sizeof(float));
  float* u    = (float*)carve(NN * HH * sizeof(float));
  float* pagg = (float*)carve((size_t)NN * NTJ * HH * sizeof(float));
  float* pcrd = (float*)carve((size_t)NN * NTJ * 3 * sizeof(float));
  _Float16* wProj = (_Float16*)carve(64 * 128 * sizeof(_Float16));
  _Float16 *wWa[LL], *wWb[LL], *wWc[LL], *wWe2[LL], *wWc1[LL], *wWn1[LL], *wWn2[LL];
  for (int l = 0; l < LL; ++l) {
    wWa[l]  = (_Float16*)carve(128 * 128 * sizeof(_Float16));
    wWb[l]  = (_Float16*)carve(128 * 128 * sizeof(_Float16));
    wWc[l]  = (_Float16*)carve(32  * 128 * sizeof(_Float16));
    wWe2[l] = (_Float16*)carve(128 * 128 * sizeof(_Float16));
    wWc1[l] = (_Float16*)carve(128 * 128 * sizeof(_Float16));
    wWn1[l] = (_Float16*)carve(256 * 128 * sizeof(_Float16));
    wWn2[l] = (_Float16*)carve(128 * 128 * sizeof(_Float16));
  }

  auto prep = [&](const float* src, int srcK, _Float16* dst, int Kpad) {
    int total = Kpad * 128;
    prep_b_kernel<<<(total + 255) / 256, 256, 0, stream>>>(src, srcK, 128, dst, Kpad, 128);
  };
  prep(proj_W, 64, wProj, 64);
  for (int l = 0; l < LL; ++l) {
    const float* e1 = eW1 + (size_t)l * 272 * 128;
    prep(e1,             128, wWa[l], 128);
    prep(e1 + 128 * 128, 128, wWb[l], 128);
    prep(e1 + 256 * 128,  16, wWc[l],  32);
    prep(eW2 + (size_t)l * 128 * 128, 128, wWe2[l], 128);
    prep(cW1 + (size_t)l * 128 * 128, 128, wWc1[l], 128);
    prep(nW1 + (size_t)l * 256 * 128, 256, wWn1[l], 256);
    prep(nW2 + (size_t)l * 128 * 128, 128, wWn2[l], 128);
  }

  center_kernel<<<1, NN, 0, stream>>>(anchor, xw);

  const int gemm_grid = (NN / 16) * (HH / 16);  // 192
  gemm16_kernel<<<gemm_grid, 32, 0, stream>>>(z, nullptr, 64, 64, 64,
                                              wProj, proj_b, nullptr, h, HH, 0);

  for (int l = 0; l < LL; ++l) {
    gemm16_kernel<<<gemm_grid, 32, 0, stream>>>(h, nullptr, 128, 128, 128,
                                                wWa[l], eb1 + l * HH, nullptr, Pm, HH, 0);
    gemm16_kernel<<<gemm_grid, 32, 0, stream>>>(h, nullptr, 128, 128, 128,
                                                wWb[l], nullptr, nullptr, Qm, HH, 0);
    edge_kernel<<<NN, 32 * WPB, 0, stream>>>(xw, Pm, Qm, wWc[l], wWe2[l], wWc1[l],
                                             eb2 + l * HH, cb1 + l * HH, cW2 + l * HH,
                                             pagg, pcrd);
    reduce_kernel<<<(NN * HH + NN * 3 + 255) / 256, 256, 0, stream>>>(pagg, pcrd, agg, xw);
    gemm16_kernel<<<gemm_grid, 32, 0, stream>>>(h, agg, 128, 256, 128,
                                                wWn1[l], nb1 + l * HH, nullptr, u, HH, 1);
    gemm16_kernel<<<gemm_grid, 32, 0, stream>>>(u, nullptr, 128, 128, 128,
                                                wWn2[l], nb2 + l * HH, h, h, HH, 0);
  }

  (void)hipMemcpyAsync(d_out, xw, NN * 3 * sizeof(float),
                       hipMemcpyDeviceToDevice, stream);
}
